// Attention_47717086658562
// MI455X (gfx1250) — matrix operand order repaired and checked by reading, not compiled, not run on hardware
//
#include <hip/hip_runtime.h>
#include <math.h>

// ---------------------------------------------------------------------------
// Shapes
// ---------------------------------------------------------------------------
#define BZ   256
#define SRCN 1024
#define DIM  512
#define ATTN_ELEMS (256UL * 512UL * 1024UL)   // (B, D, S)

// Tiling for the fused GEMM kernel
#define BM 128          // s rows per block (resident in LDS, full K)
#define BK 32           // K per WMMA step
#define LDS_STRIDE 520  // 512 + 8 bf16 pad (keeps 16B alignment, staggers banks)
#define SB_BYTES (BM * LDS_STRIDE * 2)        // 133120 B src tile
#define SMEM_BYTES (SB_BYTES + 512*4*3 + 128*4)

typedef __attribute__((ext_vector_type(16))) __bf16          v16bf;
typedef __attribute__((ext_vector_type(8)))  float           v8f;
typedef __attribute__((ext_vector_type(8)))  unsigned short  us8;  // 16 B
typedef __attribute__((ext_vector_type(4)))  unsigned short  us4;  // 8 B

union FragU { v16bf v; us8 h[2]; };

static __device__ __forceinline__ unsigned short f32_to_bf16(float f) {
  unsigned int u = __float_as_uint(f);
  u += 0x7FFFu + ((u >> 16) & 1u);   // round-to-nearest-even
  return (unsigned short)(u >> 16);
}

// ---------------------------------------------------------------------------
// Kernel 1: pack W_ref (rows 0..511) and conv_w (rows 512..1023) as bf16
// ---------------------------------------------------------------------------
__global__ void prep_weights(const float* __restrict__ wref,
                             const float* __restrict__ convw,
                             unsigned short* __restrict__ wcat) {
  int i = blockIdx.x * 256 + threadIdx.x;          // 0 .. 1024*512-1
  float val = (i < DIM * DIM) ? wref[i] : convw[i - DIM * DIM];
  wcat[i] = f32_to_bf16(val);
}

// ---------------------------------------------------------------------------
// Kernel 2: q[b,e] = sum_d tgt[b,d] * W_q[e,d]   (tiny: 67 MFLOP)
// ---------------------------------------------------------------------------
__global__ void q_kernel(const float* __restrict__ tgt,
                         const float* __restrict__ wq,
                         float* __restrict__ qbuf) {
  __shared__ float t[DIM];
  int b = blockIdx.x;
  for (int i = threadIdx.x; i < DIM; i += 256) t[i] = tgt[b * DIM + i];
  __syncthreads();
  int wave = threadIdx.x >> 5, lane = threadIdx.x & 31;
  for (int e = wave; e < DIM; e += 8) {
    float s = 0.f;
    #pragma unroll
    for (int i = 0; i < DIM / 32; ++i) {           // coalesced row read
      int d = i * 32 + lane;
      s += wq[e * DIM + d] * t[d];
    }
    #pragma unroll
    for (int o = 16; o > 0; o >>= 1) s += __shfl_down(s, o, 32);
    if (lane == 0) qbuf[b * DIM + e] = s;
  }
}

// ---------------------------------------------------------------------------
// Kernel 3: fused dual GEMM per (b, s-tile).
//   C[M=weight_row, N=s] = w_cat(1024x512,bf16) x src_b^T(512 x 128,bf16)
//   rows 0..511  -> r  -> u[s] += v[e]*tanh(q[e]+r)   (lane-local + ds_add_f32)
//   rows 512..1023 -> attn_h[b,o,s] = val + conv_b[o] (coalesced along s)
// src tile is converted f32->bf16 once and kept RESIDENT in 130KB of LDS.
// ---------------------------------------------------------------------------
__global__ void main_kernel(const float* __restrict__ src,
                            const unsigned short* __restrict__ wcat,
                            const float* __restrict__ qbuf,
                            const float* __restrict__ vvec,
                            const float* __restrict__ convb,
                            float* __restrict__ attn,
                            float* __restrict__ ubuf) {
  extern __shared__ unsigned char smem_raw[];
  unsigned short* sB = (unsigned short*)smem_raw;                 // [BM][LDS_STRIDE]
  float* sQ  = (float*)(smem_raw + SB_BYTES);                     // 512
  float* sV  = sQ + DIM;                                          // 512
  float* sCB = sV + DIM;                                          // 512
  float* sU  = sCB + DIM;                                         // 128

  const int b  = blockIdx.x >> 3;        // batch
  const int st = blockIdx.x & 7;         // s tile
  const int s0 = st * BM;
  const float* srcB = src + ((size_t)b * SRCN) * DIM;

  // ---- stage src tile as bf16 into LDS (read exactly once from HBM) ----
  for (int idx = threadIdx.x; idx < BM * (DIM / 4); idx += 256) {
    int r  = idx >> 7;                   // row 0..127
    int c4 = idx & 127;                  // float4 column
    float4 f = ((const float4*)(srcB + (size_t)(s0 + r) * DIM))[c4];
    us4 pk;
    pk[0] = f32_to_bf16(f.x); pk[1] = f32_to_bf16(f.y);
    pk[2] = f32_to_bf16(f.z); pk[3] = f32_to_bf16(f.w);
    *(us4*)&sB[r * LDS_STRIDE + c4 * 4] = pk;
  }
  for (int i = threadIdx.x; i < DIM; i += 256) {
    sQ[i]  = qbuf[b * DIM + i];
    sV[i]  = vvec[i];
    sCB[i] = convb[i];
  }
  for (int i = threadIdx.x; i < BM; i += 256) sU[i] = 0.f;
  __syncthreads();

  const int wave = threadIdx.x >> 5;
  const int lane = threadIdx.x & 31;
  const int wm = wave >> 1;              // 0..3 : 32 weight rows each
  const int wn = wave & 1;               // 0..1 : 64 s columns each
  const int lh = lane & 15;
  const bool hi = lane >= 16;

  for (int nt = 0; nt < 8; ++nt) {
    const int nbase = nt * 128;          // combined weight-row tile base
    v8f acc[2][4];
    #pragma unroll
    for (int mi = 0; mi < 2; ++mi)
      #pragma unroll
      for (int ni = 0; ni < 4; ++ni) {
        v8f z = {0.f,0.f,0.f,0.f,0.f,0.f,0.f,0.f};
        acc[mi][ni] = z;
      }

    for (int kc = 0; kc < DIM / BK; ++kc) {
      const int k0 = kc * BK;
      // A fragments: weights from global (bf16, hot in L2). 16x32 layout:
      // elems 0..7  -> K = k0 + (hi?8:0)  + 0..7
      // elems 8..15 -> K = k0 + 16 + (hi?8:0) + 0..7
      v16bf afr[2];
      #pragma unroll
      for (int mi = 0; mi < 2; ++mi) {
        int row = nbase + wm * 32 + mi * 16 + lh;
        const unsigned short* p = wcat + (size_t)row * DIM + k0 + (hi ? 8 : 0);
        FragU f;
        f.h[0] = *(const us8*)p;
        f.h[1] = *(const us8*)(p + 16);
        afr[mi] = f.v;
        __builtin_prefetch(p + BK, 0, 0);          // global_prefetch next K chunk
      }
      // B fragments: resident src tile in LDS. 32x16 layout:
      // elems 0..7 -> K = k0 + (hi?8:0) + 0..7 ; elems 8..15 -> K = k0 + 16 + (hi?8:0) + 0..7
      v16bf bfr[4];
      #pragma unroll
      for (int ni = 0; ni < 4; ++ni) {
        int sl = wn * 64 + ni * 16 + lh;
        const unsigned short* p = &sB[sl * LDS_STRIDE + k0 + (hi ? 8 : 0)];
        FragU f;
        f.h[0] = *(const us8*)p;
        f.h[1] = *(const us8*)(p + 16);
        bfr[ni] = f.v;
      }
      #pragma unroll
      for (int mi = 0; mi < 2; ++mi)
        #pragma unroll
        for (int ni = 0; ni < 4; ++ni)
          acc[mi][ni] = __builtin_amdgcn_wmma_f32_16x16x32_bf16(
              false, afr[mi], false, bfr[ni], (short)0, acc[mi][ni],
              false, false);
    }

    // ---- epilogues ----
    if (nt < 4) {
      // r part: u[s] += v[e] * tanh(q[e] + r[e,s]); lane owns fixed s per ni
      #pragma unroll
      for (int ni = 0; ni < 4; ++ni) {
        float part = 0.f;
        int sl = wn * 64 + ni * 16 + lh;
        #pragma unroll
        for (int mi = 0; mi < 2; ++mi) {
          int ebase = nbase + wm * 32 + mi * 16 + (hi ? 8 : 0);
          #pragma unroll
          for (int j = 0; j < 8; ++j) {
            int e = ebase + j;
            part += sV[e] * tanhf(sQ[e] + acc[mi][ni][j]);
          }
        }
        atomicAdd(&sU[sl], part);                  // ds_add_f32
      }
    } else {
      // conv part: attn[b, o, s0+sl] = val + conv_b[o]; lanes => contiguous s
      #pragma unroll
      for (int ni = 0; ni < 4; ++ni) {
        int sl = wn * 64 + ni * 16 + lh;
        size_t base = (size_t)b * (DIM * (size_t)SRCN) + (size_t)(s0 + sl);
        #pragma unroll
        for (int mi = 0; mi < 2; ++mi) {
          int obase = (nbase - 512) + wm * 32 + mi * 16 + (hi ? 8 : 0);
          #pragma unroll
          for (int j = 0; j < 8; ++j) {
            int o = obase + j;
            attn[base + (size_t)o * SRCN] = acc[mi][ni][j] + sCB[o];
          }
        }
      }
    }
  }

  __syncthreads();
  for (int i = threadIdx.x; i < BM; i += 256)
    ubuf[b * SRCN + s0 + i] = sU[i];
}

// ---------------------------------------------------------------------------
// Kernel 4: masked softmax over S per batch + mask output
// ---------------------------------------------------------------------------
__global__ void softmax_kernel(const float* __restrict__ ubuf,
                               const unsigned char* __restrict__ mask_in,
                               const int* __restrict__ prev,
                               float* __restrict__ logits,
                               float* __restrict__ mask_out) {
  __shared__ float red[8];
  const int b = blockIdx.x;
  const int pi = prev[b];
  const int wave = threadIdx.x >> 5, lane = threadIdx.x & 31;

  float sc[4], mk[4];
  float lmax = -__builtin_inff();
  #pragma unroll
  for (int i = 0; i < 4; ++i) {
    int s = i * 256 + threadIdx.x;
    bool m = (mask_in[b * SRCN + s] != 0) || (s == pi);
    float val = 10.0f * tanhf(ubuf[b * SRCN + s]);
    sc[i] = m ? -__builtin_inff() : val;
    mk[i] = m ? 1.0f : 0.0f;
    lmax = fmaxf(lmax, sc[i]);
  }
  #pragma unroll
  for (int o = 16; o > 0; o >>= 1) lmax = fmaxf(lmax, __shfl_down(lmax, o, 32));
  if (lane == 0) red[wave] = lmax;
  __syncthreads();
  if (threadIdx.x == 0) {
    float m = red[0];
    #pragma unroll
    for (int i = 1; i < 8; ++i) m = fmaxf(m, red[i]);
    red[0] = m;
  }
  __syncthreads();
  float bmax = red[0];
  __syncthreads();

  float ev[4];
  float lsum = 0.f;
  #pragma unroll
  for (int i = 0; i < 4; ++i) { ev[i] = expf(sc[i] - bmax); lsum += ev[i]; }
  #pragma unroll
  for (int o = 16; o > 0; o >>= 1) lsum += __shfl_down(lsum, o, 32);
  if (lane == 0) red[wave] = lsum;
  __syncthreads();
  if (threadIdx.x == 0) {
    float m = 0.f;
    #pragma unroll
    for (int i = 0; i < 8; ++i) m += red[i];
    red[0] = m;
  }
  __syncthreads();
  float inv = 1.0f / red[0];
  #pragma unroll
  for (int i = 0; i < 4; ++i) {
    int s = i * 256 + threadIdx.x;
    logits[b * SRCN + s]   = ev[i] * inv;
    mask_out[b * SRCN + s] = mk[i];
  }
}

// ---------------------------------------------------------------------------
// Launch
// ---------------------------------------------------------------------------
extern "C" void kernel_launch(void* const* d_in, const int* in_sizes, int n_in,
                              void* d_out, int out_size, void* d_ws, size_t ws_size,
                              hipStream_t stream) {
  const float*         src   = (const float*)d_in[0];
  const float*         tgt   = (const float*)d_in[1];
  const unsigned char* mask  = (const unsigned char*)d_in[2];
  const int*           prev  = (const int*)d_in[3];
  const float*         wq    = (const float*)d_in[4];
  const float*         wref  = (const float*)d_in[5];
  const float*         vvec  = (const float*)d_in[6];
  const float*         convw = (const float*)d_in[7];
  const float*         convb = (const float*)d_in[8];

  // workspace: [w_cat bf16 1MB][q f32 512KB][u f32 1MB]
  unsigned short* wcat = (unsigned short*)d_ws;
  float* qbuf = (float*)((char*)d_ws + 1024UL * 512UL * 2UL);
  float* ubuf = (float*)((char*)d_ws + 1024UL * 512UL * 2UL + 256UL * 512UL * 4UL);

  float* attn    = (float*)d_out;
  float* logits  = (float*)d_out + ATTN_ELEMS;
  float* maskout = logits + 256UL * 1024UL;

  prep_weights<<<(1024 * 512) / 256, 256, 0, stream>>>(wref, convw, wcat);
  q_kernel<<<BZ, 256, 0, stream>>>(tgt, wq, qbuf);
  main_kernel<<<BZ * (SRCN / BM), 256, SMEM_BYTES, stream>>>(
      src, wcat, qbuf, vvec, convb, attn, ubuf);
  softmax_kernel<<<BZ, 256, 0, stream>>>(ubuf, mask, prev, logits, maskout);
}
